// GPTQMarlinMoE_42348377539245
// MI455X (gfx1250) — compile-verified, matrix operand order checked by reading, not run on hardware
//
#include <hip/hip_runtime.h>
#include <cstdint>
#include <cmath>

// ---------------- problem constants (from reference) ----------------
constexpr int E    = 8;
constexpr int T    = 1024;
constexpr int D    = 1024;
constexpr int IDIM = 5632;          // I
constexpr int G    = 128;           // quant group size (== K-chunk)
constexpr int LDA  = 136;           // padded LDS stride (bf16 elems), 16B-aligned rows

typedef __attribute__((ext_vector_type(16))) __bf16 v16bf;
typedef __attribute__((ext_vector_type(8)))  float  v8f;
typedef __attribute__((ext_vector_type(4)))  int    iv4;     // NT weight loads
typedef __attribute__((ext_vector_type(4)))  unsigned int uv4; // TDM group 0
typedef __attribute__((ext_vector_type(8)))  int    iv8;     // TDM group 1

// ---------------- helpers ----------------
__device__ __forceinline__ unsigned short f2bf(float f) {
    unsigned u = __float_as_uint(f);
    unsigned r = (u + 0x7FFFu + ((u >> 16) & 1u)) >> 16;   // RNE
    return (unsigned short)r;
}
__device__ __forceinline__ unsigned pk2(unsigned short lo, unsigned short hi) {
    return (unsigned)lo | ((unsigned)hi << 16);
}

// Load one 16x32 bf16 fragment (A-style: lanes = M rows, K packed 2/VGPR,
// half-wave K-split per ISA 7.12.2). B fragments use the mirrored layout:
// tile stored [n][k], lane = N column. Lowers to 2x ds_load_b128.
__device__ __forceinline__ v16bf load_frag(const unsigned short* tile, int lane) {
    const int half = lane >> 4;
    const int m    = lane & 15;
    const unsigned short* p = tile + m * LDA + (half << 3);
    union { v16bf v; unsigned u[8]; } f;
#pragma unroll
    for (int i = 0; i < 8; ++i) {
        const int k0 = ((i & 4) << 2) + ((i & 3) << 1); // 0,2,4,6,16,18,20,22
        f.u[i] = *(const unsigned*)(p + k0);
    }
    return f.v;
}

__device__ __forceinline__ v8f wmma_bf16(v16bf a, v16bf b, v8f c) {
    return __builtin_amdgcn_wmma_f32_16x16x32_bf16(false, a, false, b,
                                                   (short)0, c, false, false);
}

// TDM: async 2D bf16 tile load (256 rows x 128 elems, row stride IDIM elems)
// into LDS with the padded LDA=136 layout (pad 4 DWORDs every 64 DWORDs).
// tensor_dim1 = rows_valid gives hardware zero-fill for the row tail.
__device__ __forceinline__ void tdm_load_tile(void* lds_dst, const void* gsrc,
                                              int rows_valid) {
    const unsigned lw = (unsigned)(uintptr_t)lds_dst;          // LDS byte address
    const unsigned long long ga = (unsigned long long)(uintptr_t)gsrc;
    uv4 g0 = { 1u,                                   // count=1, user descriptor
               lw,                                   // lds_addr
               (unsigned)ga,                         // global_addr[31:0]
               (unsigned)((ga >> 32) & 0x01FFFFFFull) | (2u << 30) }; // [56:32] | type=2
    iv8 g1 = { (int)0x07510000u,        // data_size=2B, pad_en, pad_interval=5, pad_amount=3
               (int)(128u << 16),       // tensor_dim0 = 128
               (int)((unsigned)rows_valid << 16),    // tensor_dim1 = valid rows
               (int)(128u << 16),       // tile_dim0 = 128
               256,                     // tile_dim1 = 256 (tile_dim2 = 0)
               IDIM,                    // tensor_dim0_stride[31:0]
               0, 0 };
    iv4 z4 = { 0, 0, 0, 0 };
    iv8 z8 = { 0, 0, 0, 0, 0, 0, 0, 0 };
    __builtin_amdgcn_tensor_load_to_lds(g0, g1, z4, z4, z8, 0);  // 6-arg toolchain form
}

// ---------------- kernel 1: routing (softmax + top-2 + gather) ----------------
__global__ __launch_bounds__(256) void moe_route(const float* __restrict__ gate,
                                                 int* __restrict__ cnt,
                                                 int* __restrict__ tok,
                                                 float* __restrict__ tw) {
    int t = blockIdx.x * 256 + threadIdx.x;
    if (t >= T) return;
    float p[E];
    float mx = -1e30f;
#pragma unroll
    for (int e = 0; e < E; ++e) { p[e] = gate[t * E + e]; mx = fmaxf(mx, p[e]); }
#pragma unroll
    for (int e = 0; e < E; ++e) p[e] = __expf(p[e] - mx);
    int i1 = 0;
#pragma unroll
    for (int e = 1; e < E; ++e) if (p[e] > p[i1]) i1 = e;
    int i2 = (i1 == 0) ? 1 : 0;
#pragma unroll
    for (int e = 0; e < E; ++e) if (e != i1 && p[e] > p[i2]) i2 = e;
    float denom = p[i1] + p[i2];
    float w1v = p[i1] / denom, w2v = p[i2] / denom;
    int s1 = atomicAdd(&cnt[i1], 1); tok[i1 * T + s1] = t; tw[i1 * T + s1] = w1v;
    int s2 = atomicAdd(&cnt[i2], 1); tok[i2 * T + s2] = t; tw[i2 * T + s2] = w2v;
}

// ---------------- kernel 2: x @ dequant(W1), fused SiLU*up -> act (bf16) ------
// block tile: 256 gathered rows x (64 gate cols + 64 up cols), K chunked by 128
__global__ __launch_bounds__(256) void moe_gemm1(const float* __restrict__ x,
                                                 const int* __restrict__ w1q,
                                                 const float* __restrict__ w1s,
                                                 const int* __restrict__ cnt,
                                                 const int* __restrict__ tok,
                                                 unsigned short* __restrict__ act) {
    __shared__ unsigned short sA[256 * LDA];
    __shared__ unsigned short sB[2][64 * LDA];

    const int e    = blockIdx.z;
    const int row0 = blockIdx.y * 256;
    const int c0   = blockIdx.x * 64;
    const int n_rows = cnt[e];
    if (row0 >= n_rows) return;
    const int rows_here = min(256, n_rows - row0);

    const int tidx = threadIdx.x;
    const int lane = tidx & 31;
    const int wave = tidx >> 5;
    const int wr   = wave * 32;
    const size_t twoI = 2 * (size_t)IDIM;

    const v8f vzero = {0.f, 0.f, 0.f, 0.f, 0.f, 0.f, 0.f, 0.f};
    v8f accg[2][4], accu[2][4];
#pragma unroll
    for (int i = 0; i < 2; ++i)
#pragma unroll
        for (int j = 0; j < 4; ++j) { accg[i][j] = vzero; accu[i][j] = vzero; }

    for (int kc = 0; kc < D / 128; ++kc) {
        // ---- stage A chunk: 256 rows x 128 K, fp32 -> bf16 (x is L2-resident) ----
        for (int it = 0; it < 32; ++it) {
            int lin = tidx + it * 256;
            int r   = lin >> 5;
            int k   = (lin & 31) << 2;
            float4 xv = {0.f, 0.f, 0.f, 0.f};
            if (r < rows_here) {
                int tk = tok[e * T + row0 + r];
                xv = *(const float4*)(x + (size_t)tk * D + kc * 128 + k);
            }
            uint2 v;
            v.x = pk2(f2bf(xv.x), f2bf(xv.y));
            v.y = pk2(f2bf(xv.z), f2bf(xv.w));
            *(uint2*)(&sA[r * LDA + k]) = v;
        }
        // ---- stage B chunks (gate m=0, up m=1): NT-stream int4 -> bf16, [n][k] ----
#pragma unroll
        for (int m = 0; m < 2; ++m) {
            const size_t colbase = (size_t)c0 + (size_t)m * IDIM;
            for (int it = 0; it < 8; ++it) {
                int lin = tidx + it * 256;       // 0..2047
                int k   = lin >> 4;              // 0..127
                int n4  = (lin & 15) << 2;       // 0..60
                const iv4 q = __builtin_nontemporal_load((const iv4*)(w1q +
                    ((size_t)e * D + (size_t)kc * 128 + k) * twoI + colbase + n4));
                const float4 sc = *(const float4*)(w1s +
                    ((size_t)e * (D / G) + kc) * twoI + colbase + n4);
                sB[m][(n4 + 0) * LDA + k] = f2bf((float)(q[0] - 8) * sc.x);
                sB[m][(n4 + 1) * LDA + k] = f2bf((float)(q[1] - 8) * sc.y);
                sB[m][(n4 + 2) * LDA + k] = f2bf((float)(q[2] - 8) * sc.z);
                sB[m][(n4 + 3) * LDA + k] = f2bf((float)(q[3] - 8) * sc.w);
            }
        }
        if (kc + 1 < D / 128)   // prefetch next weight chunk (global_prefetch_b8)
            __builtin_prefetch(w1q + ((size_t)e * D + (size_t)(kc + 1) * 128 +
                                      (tidx >> 1)) * twoI + c0, 0, 0);
        __syncthreads();

        // ---- WMMA over the chunk: 4 k-steps of 32 ----
#pragma unroll
        for (int ks = 0; ks < 4; ++ks) {
            v16bf a0 = load_frag(&sA[(wr +  0) * LDA + ks * 32], lane);
            v16bf a1 = load_frag(&sA[(wr + 16) * LDA + ks * 32], lane);
#pragma unroll
            for (int ct = 0; ct < 4; ++ct) {
                v16bf bg = load_frag(&sB[0][(ct * 16) * LDA + ks * 32], lane);
                v16bf bu = load_frag(&sB[1][(ct * 16) * LDA + ks * 32], lane);
                accg[0][ct] = wmma_bf16(a0, bg, accg[0][ct]);
                accg[1][ct] = wmma_bf16(a1, bg, accg[1][ct]);
                accu[0][ct] = wmma_bf16(a0, bu, accu[0][ct]);
                accu[1][ct] = wmma_bf16(a1, bu, accu[1][ct]);
            }
        }
        __syncthreads();
    }

    // ---- epilogue: act = silu(gate) * up, bf16 store ----
    const int half = lane >> 4;
    const int nl   = lane & 15;
#pragma unroll
    for (int rt = 0; rt < 2; ++rt) {
#pragma unroll
        for (int j = 0; j < 8; ++j) {
            int rl = wr + rt * 16 + j + 8 * half;     // C layout: m = j + 8*half
            if (rl >= rows_here) continue;
            size_t rowoff = ((size_t)e * T + row0 + rl) * (size_t)IDIM;
#pragma unroll
            for (int ct = 0; ct < 4; ++ct) {
                float g = accg[rt][ct][j];
                float u = accu[rt][ct][j];
                float a = (g / (1.f + __expf(-g))) * u;
                act[rowoff + c0 + ct * 16 + nl] = f2bf(a);
            }
        }
    }
}

// ---------------- kernel 3: act @ dequant(W2), weighted scatter-add ----------
// block tile: 256 gathered rows x 128 D cols, K = I chunked by 128 (44 chunks)
// A tile staged by the Tensor Data Mover (async DMA, zero-fill row tail).
__global__ __launch_bounds__(256) void moe_gemm2(const unsigned short* __restrict__ act,
                                                 const int* __restrict__ w2q,
                                                 const float* __restrict__ w2s,
                                                 const int* __restrict__ cnt,
                                                 const int* __restrict__ tok,
                                                 const float* __restrict__ tw,
                                                 float* __restrict__ out) {
    __shared__ unsigned short sA[256 * LDA];
    __shared__ unsigned short sB[128 * LDA];

    const int e    = blockIdx.z;
    const int row0 = blockIdx.y * 256;
    const int cb   = blockIdx.x * 128;
    const int n_rows = cnt[e];
    if (row0 >= n_rows) return;
    const int rows_here = min(256, n_rows - row0);

    const int tidx = threadIdx.x;
    const int lane = tidx & 31;
    const int wave = tidx >> 5;
    const int wr   = wave * 32;

    const v8f vzero = {0.f, 0.f, 0.f, 0.f, 0.f, 0.f, 0.f, 0.f};
    v8f acc[2][8];
#pragma unroll
    for (int i = 0; i < 2; ++i)
#pragma unroll
        for (int j = 0; j < 8; ++j) acc[i][j] = vzero;

    for (int kc = 0; kc < IDIM / 128; ++kc) {
        // ---- stage A chunk via TDM: 256x128 bf16 tile, row stride IDIM ----
        if (wave == 0)
            tdm_load_tile(sA,
                          act + ((size_t)e * T + row0) * (size_t)IDIM + (size_t)kc * 128,
                          rows_here);
        // ---- stage B chunk: NT-stream dequant int4 -> bf16, [n][k] ----
        for (int it = 0; it < 16; ++it) {
            int lin = tidx + it * 256;           // 0..4095
            int k   = lin >> 5;                  // 0..127
            int n4  = (lin & 31) << 2;           // 0..124
            const iv4 q = __builtin_nontemporal_load((const iv4*)(w2q +
                ((size_t)e * IDIM + (size_t)kc * 128 + k) * D + cb + n4));
            const float4 sc = *(const float4*)(w2s +
                ((size_t)e * (IDIM / G) + kc) * D + cb + n4);
            sB[(n4 + 0) * LDA + k] = f2bf((float)(q[0] - 8) * sc.x);
            sB[(n4 + 1) * LDA + k] = f2bf((float)(q[1] - 8) * sc.y);
            sB[(n4 + 2) * LDA + k] = f2bf((float)(q[2] - 8) * sc.z);
            sB[(n4 + 3) * LDA + k] = f2bf((float)(q[3] - 8) * sc.w);
        }
        if (kc + 1 < IDIM / 128)
            __builtin_prefetch(w2q + ((size_t)e * IDIM + (size_t)(kc + 1) * 128 +
                                      (tidx >> 1)) * D + cb, 0, 0);
        if (wave == 0)
            __builtin_amdgcn_s_wait_tensorcnt(0);   // TDM tile landed
        __syncthreads();

#pragma unroll
        for (int ks = 0; ks < 4; ++ks) {
            v16bf a0 = load_frag(&sA[(wr +  0) * LDA + ks * 32], lane);
            v16bf a1 = load_frag(&sA[(wr + 16) * LDA + ks * 32], lane);
#pragma unroll
            for (int ct = 0; ct < 8; ++ct) {
                v16bf b = load_frag(&sB[(ct * 16) * LDA + ks * 32], lane);
                acc[0][ct] = wmma_bf16(a0, b, acc[0][ct]);
                acc[1][ct] = wmma_bf16(a1, b, acc[1][ct]);
            }
        }
        __syncthreads();
    }

    // ---- epilogue: out[t] += tw * y (each token has exactly K=2 contributions) ----
    const int half = lane >> 4;
    const int nl   = lane & 15;
#pragma unroll
    for (int rt = 0; rt < 2; ++rt) {
#pragma unroll
        for (int j = 0; j < 8; ++j) {
            int rl = wr + rt * 16 + j + 8 * half;
            if (rl >= rows_here) continue;
            int   tk  = tok[e * T + row0 + rl];
            float wgt = tw[e * T + row0 + rl];
            float* orow = out + (size_t)tk * D + cb;
#pragma unroll
            for (int ct = 0; ct < 8; ++ct)
                unsafeAtomicAdd(&orow[ct * 16 + nl], wgt * acc[rt][ct][j]);
        }
    }
}

// ---------------- host-side launch ----------------
extern "C" void kernel_launch(void* const* d_in, const int* in_sizes, int n_in,
                              void* d_out, int out_size, void* d_ws, size_t ws_size,
                              hipStream_t stream) {
    (void)in_sizes; (void)n_in; (void)out_size; (void)ws_size;
    const float* x      = (const float*)d_in[0];
    const float* gating = (const float*)d_in[1];
    const int*   w1q    = (const int*)d_in[2];
    const int*   w2q    = (const int*)d_in[3];
    const float* w1s    = (const float*)d_in[4];
    const float* w2s    = (const float*)d_in[5];
    float* out = (float*)d_out;

    // workspace layout
    char* p = (char*)d_ws;
    int*   cnt = (int*)p;                                        // E ints (256B reserved)
    int*   tok = (int*)(p + 256);                                // E*T ints
    float* tw  = (float*)(p + 256 + (size_t)E * T * 4);          // E*T floats
    unsigned short* act = (unsigned short*)(p + 256 + 2 * (size_t)E * T * 4); // E*T*I bf16

    (void)hipMemsetAsync(cnt, 0, 256, stream);
    (void)hipMemsetAsync(out, 0, (size_t)T * D * sizeof(float), stream);

    moe_route<<<T / 256, 256, 0, stream>>>(gating, cnt, tok, tw);
    moe_gemm1<<<dim3(IDIM / 64, T / 256, E), 256, 0, stream>>>(x, w1q, w1s, cnt, tok, act);
    moe_gemm2<<<dim3(D / 128, T / 256, E), 256, 0, stream>>>(act, w2q, w2s, cnt, tok, tw, out);
}
